// DecoderFactoredLSTMAtt_42717744726314
// MI455X (gfx1250) — matrix operand order repaired and checked by reading, not compiled, not run on hardware
//
#include <hip/hip_runtime.h>

// ---------------- problem dimensions ----------------
constexpr int B_     = 64;
constexpr int T_     = 40;
constexpr int P_     = 196;
constexpr int FEAT_  = 2048;
constexpr int E_     = 512;
constexpr int F_     = 512;
constexpr int H_     = 1024;
constexpr int ATT_   = 512;
constexpr int VOCAB_ = 20000;
constexpr int IN_    = E_ + FEAT_;     // 2560
constexpr int VOCABP_= 20096;          // VOCAB padded to TN multiple (157*128)

// ---------------- vector types ----------------
typedef __attribute__((ext_vector_type(16))) _Float16 v16h;
typedef __attribute__((ext_vector_type(8)))  _Float16 v8h;
typedef __attribute__((ext_vector_type(4)))  _Float16 v4h;
typedef __attribute__((ext_vector_type(2)))  _Float16 v2h;
typedef __attribute__((ext_vector_type(8)))  float    v8f;

// ---------------- WMMA GEMM: C[M,N] = act(A[M,K] @ B[K,ldb] + bias (+C)) ----
// A,B row-major f16. Block = 128 threads = 4 waves computing a 32x128 C tile:
// each wave owns two 16-wide N slices x two 16x16 M tiles -> 4 WMMAs per
// K-step; A and B fragments each reused by 2 MMAs. K stepped by 32. LDS is
// double-buffered (stage tile k+1 while computing tile k; ONE barrier per
// K-step). B is transposed in LDS so all fragments are 16-byte ds_load_b128;
// staging pairs adjacent K rows into 4-byte ds_store_b32.
// Requirements: M%32==0, K%32==0, ldb%TN==0 and ldb>=N (pad B with zero cols).
constexpr int TM = 32, TN = 128, TK = 32;
constexpr int LDS_A = TK + 8;   // 40 halves -> 80B rows: 16B-aligned, staggered banks
constexpr int LDS_B = TK + 8;

template<int ACT, bool ACCUM, bool WRITE_HALF>
__global__ __launch_bounds__(128)
void gemm_wmma_f16(const _Float16* __restrict__ A, long sAz,
                   const _Float16* __restrict__ Bm, long sBz,
                   const float*    __restrict__ bias, long sBiasz,
                   float*          __restrict__ C, long sCz,
                   _Float16*       __restrict__ Ch, long sChz,
                   int M, int N, int K, int ldb)
{
  __shared__ _Float16 As[2][TM * LDS_A];   // 32 x 32 (+pad), ping-pong
  __shared__ _Float16 Bs[2][TN * LDS_B];   // transposed tile Bs[n][k], ping-pong

  const int z = blockIdx.z;
  A    += (long)z * sAz;
  Bm   += (long)z * sBz;
  C    += (long)z * sCz;
  bias += (long)z * sBiasz;
  if constexpr (WRITE_HALF) Ch += (long)z * sChz;

  const int tid  = threadIdx.x;
  const int lane = tid & 31;
  const int wave = tid >> 5;
  const int m0 = blockIdx.y * TM;
  const int n0 = blockIdx.x * TN;

  v8f acc[2][2] = {};   // [m-tile][n-tile]

  const int ar  = tid >> 2;          // A staging: row 0..31
  const int ac  = (tid & 3) * 8;     // A staging: col 0,8,16,24  (v8h = 16B)
  const int bc  = (tid & 31) * 4;    // B staging: local col 0,4..124
  const int rp2 = (tid >> 5) * 2;    // B staging: even base row 0,2,4,6

  // staging base pointers, advanced by TK (A) / TK*ldb (B) per K-step
  const _Float16* aptr = A + (long)(m0 + ar) * K + ac;
  const _Float16* bptr = Bm + (long)rp2 * ldb + (n0 + bc);

  // stage one 32xTK A tile + TKx128 B tile (transposed) into buffer sb
  auto stage = [&](int sb, const _Float16* ap, const _Float16* bp) {
    *(v8h*)(&As[sb][ar * LDS_A + ac]) = *(const v8h*)ap;
    #pragma unroll
    for (int i = 0; i < 4; ++i) {            // row pairs {rp2, rp2+1} + 8*i
      const int r = rp2 + i * 8;
      v4h b0 = *(const v4h*)bp;
      v4h b1 = *(const v4h*)(bp + ldb);
      #pragma unroll
      for (int j = 0; j < 4; ++j) {
        v2h pk; pk[0] = b0[j]; pk[1] = b1[j];
        *(v2h*)(&Bs[sb][(bc + j) * LDS_B + r]) = pk;   // b32 store, 4B aligned
      }
      bp += 8L * ldb;
    }
  };

  stage(0, aptr, bptr);
  __syncthreads();

  int sb = 0;
  for (int k0 = 0; k0 < K; k0 += TK) {
    if (k0 + TK < K)                          // overlap next-tile staging
      stage(sb ^ 1, aptr + TK, bptr + (long)TK * ldb);

    // ---- A fragments (16x32): lane L<16 -> K-base 0, L>=16 -> K-base 8 ----
    const int am  = lane & 15;
    const int akb = (lane >> 4) << 3;          // 0 or 8
    v16h afrag[2];
    #pragma unroll
    for (int mt = 0; mt < 2; ++mt) {
      const _Float16* ap = &As[sb][(mt * 16 + am) * LDS_A + akb];
      v8h alo = *(const v8h*)ap;
      v8h ahi = *(const v8h*)(ap + 16);
      #pragma unroll
      for (int i = 0; i < 8; ++i) { afrag[mt][i] = alo[i]; afrag[mt][i + 8] = ahi[i]; }
    }
    // ---- B fragments (32x16): lane L<16 -> K-base 0, L>=16 -> K-base 8 ----
    const int bkb = (lane >> 4) << 3;          // 0 or 8
    #pragma unroll
    for (int nt = 0; nt < 2; ++nt) {
      const int bn = nt * 64 + (wave << 4) + (lane & 15);
      const _Float16* bp = &Bs[sb][bn * LDS_B + bkb];
      v8h blo = *(const v8h*)bp;
      v8h bhi = *(const v8h*)(bp + 16);
      v16h bfrag;
      #pragma unroll
      for (int i = 0; i < 8; ++i) { bfrag[i] = blo[i]; bfrag[i + 8] = bhi[i]; }
      acc[0][nt] = __builtin_amdgcn_wmma_f32_16x16x32_f16(false, afrag[0], false, bfrag,
                                                          (short)0, acc[0][nt], false, false);
      acc[1][nt] = __builtin_amdgcn_wmma_f32_16x16x32_f16(false, afrag[1], false, bfrag,
                                                          (short)0, acc[1][nt], false, false);
    }
    aptr += TK;
    bptr += (long)TK * ldb;
    sb ^= 1;
    __syncthreads();   // single barrier per K-step (ping-pong safety)
  }

  // ---- epilogue: C/D layout: VGPR r -> M = r + 8*(lane>=16), N = lane%16 ----
  const int cn = lane & 15;
  const int mb = (lane >> 4) << 3;
  #pragma unroll
  for (int nt = 0; nt < 2; ++nt) {
    const int gn = n0 + nt * 64 + (wave << 4) + cn;
    if (gn < N) {
      const float bv = bias[gn];
      #pragma unroll
      for (int mt = 0; mt < 2; ++mt) {
        #pragma unroll
        for (int r = 0; r < 8; ++r) {
          const long idx = (long)(m0 + mt * 16 + mb + r) * N + gn;
          float v = acc[mt][nt][r] + bv;
          if constexpr (ACCUM) v += C[idx];
          if constexpr (ACT == 1) v = 1.0f / (1.0f + __expf(-v));
          C[idx] = v;
          if constexpr (WRITE_HALF) Ch[idx] = (_Float16)v;
        }
      }
    }
  }
}

// ---------------- fp32 -> fp16 conversion ----------------
__global__ void f32_to_f16_kernel(const float* __restrict__ s,
                                  _Float16* __restrict__ d, long n) {
  long i = (long)blockIdx.x * blockDim.x + threadIdx.x;
  long stride = (long)gridDim.x * blockDim.x;
  for (; i < n; i += stride) d[i] = (_Float16)s[i];
}

// ---- fp32 -> fp16 with column padding: d[r][c<N]=s[r][c], d[r][c>=N]=0 ----
__global__ void f32_to_f16_pad_kernel(const float* __restrict__ s,
                                      _Float16* __restrict__ d,
                                      int rows, int N, int ldd) {
  long i = (long)blockIdx.x * blockDim.x + threadIdx.x;
  long stride = (long)gridDim.x * blockDim.x;
  const long n = (long)rows * ldd;
  for (; i < n; i += stride) {
    int c = (int)(i % ldd);
    long r = i / ldd;
    d[i] = (c < N) ? (_Float16)s[r * N + c] : (_Float16)0.0f;
  }
}

// ---------------- embedding gather: embs[T][B][E] (f16) ----------------
__global__ void gather_embs_kernel(const int* __restrict__ captions,
                                   const float* __restrict__ embed,
                                   _Float16* __restrict__ out) {
  long i = (long)blockIdx.x * blockDim.x + threadIdx.x;
  const long n = (long)T_ * B_ * E_;
  if (i >= n) return;
  int e = (int)(i % E_);
  long tb = i / E_;
  int b = (int)(tb % B_);
  int t = (int)(tb / B_);
  int tok = captions[b * T_ + t];
  out[i] = (_Float16)embed[(long)tok * E_ + e];
}

// ---------------- zero-init recurrent state ----------------
__global__ void init_state_kernel(_Float16* __restrict__ h16,
                                  float* __restrict__ c) {
  int i = blockIdx.x * blockDim.x + threadIdx.x;
  if (i < B_ * H_) { h16[i] = (_Float16)0.0f; c[i] = 0.0f; }
}

// ---------------- Bahdanau attention: scores+softmax+context -------------
// One block per batch element b.
__global__ __launch_bounds__(256)
void attention_kernel(const float* __restrict__ att1,     // [B,P,ATT]
                      const float* __restrict__ att2,     // [B,ATT]
                      const float* __restrict__ full_w,   // [ATT]
                      const float* __restrict__ full_b,   // [1]
                      const float* __restrict__ features, // [B,P,FEAT]
                      float* __restrict__ alphas,         // [B,T,P]
                      int t,
                      float* __restrict__ awe)            // [B,FEAT]
{
  const int b = blockIdx.x;
  const int tid = threadIdx.x;
  __shared__ float e_s[P_];
  __shared__ float red[256];

  const float* a2 = att2 + (long)b * ATT_;
  for (int p = tid; p < P_; p += 256) {
    const float* a1 = att1 + ((long)b * P_ + p) * ATT_;
    float s = 0.0f;
    for (int a = 0; a < ATT_; ++a) {
      float v = a1[a] + a2[a];
      s += (v > 0.0f ? v : 0.0f) * full_w[a];
    }
    e_s[p] = s + full_b[0];
  }
  __syncthreads();

  float m = -1e30f;
  for (int p = tid; p < P_; p += 256) m = fmaxf(m, e_s[p]);
  red[tid] = m; __syncthreads();
  for (int s = 128; s > 0; s >>= 1) {
    if (tid < s) red[tid] = fmaxf(red[tid], red[tid + s]);
    __syncthreads();
  }
  const float mx = red[0]; __syncthreads();

  float sum = 0.0f;
  for (int p = tid; p < P_; p += 256) {
    float ex = __expf(e_s[p] - mx);
    e_s[p] = ex; sum += ex;
  }
  red[tid] = sum; __syncthreads();
  for (int s = 128; s > 0; s >>= 1) {
    if (tid < s) red[tid] += red[tid + s];
    __syncthreads();
  }
  const float inv = 1.0f / red[0]; __syncthreads();

  for (int p = tid; p < P_; p += 256) {
    float a = e_s[p] * inv;
    e_s[p] = a;
    alphas[((long)b * T_ + t) * P_ + p] = a;
  }
  __syncthreads();

  // context: awe[b,f] = sum_p alpha[p] * features[b,p,f]
  const float* fb = features + (long)b * P_ * FEAT_;
  for (int f = tid; f < FEAT_; f += 256) {
    float s = 0.0f;
    for (int p = 0; p < P_; ++p) s += e_s[p] * fb[(long)p * FEAT_ + f];
    awe[(long)b * FEAT_ + f] = s;
  }
}

// ---------------- x = concat(emb_t, gate * awe) in f16 ----------------
__global__ void build_x_kernel(const _Float16* __restrict__ embs_t, // [B,E]
                               const float* __restrict__ gate,      // [B,FEAT]
                               const float* __restrict__ awe,       // [B,FEAT]
                               _Float16* __restrict__ x) {          // [B,IN]
  long i = (long)blockIdx.x * blockDim.x + threadIdx.x;
  if (i >= (long)B_ * IN_) return;
  int b = (int)(i / IN_);
  int j = (int)(i % IN_);
  _Float16 v;
  if (j < E_) v = embs_t[b * E_ + j];
  else {
    int f = j - E_;
    v = (_Float16)(gate[(long)b * FEAT_ + f] * awe[(long)b * FEAT_ + f]);
  }
  x[i] = v;
}

// ---------------- LSTM pointwise update ----------------
__global__ void lstm_update_kernel(const float* __restrict__ g,  // [4,B,H]
                                   float* __restrict__ c,        // [B,H]
                                   _Float16* __restrict__ h16,   // [B,H]
                                   _Float16* __restrict__ hidH)  // -> hiddens[t]
{
  int i = blockIdx.x * blockDim.x + threadIdx.x;
  const int BH = B_ * H_;
  if (i >= BH) return;
  const float gi = 1.0f / (1.0f + __expf(-g[0 * BH + i]));
  const float gf = 1.0f / (1.0f + __expf(-g[1 * BH + i]));
  const float go = 1.0f / (1.0f + __expf(-g[2 * BH + i]));
  const float gc = tanhf(g[3 * BH + i]);
  const float cn = gf * c[i] + gi * gc;
  const float hn = go * cn;   // reference applies no tanh here
  c[i] = cn;
  h16[i] = (_Float16)hn;
  hidH[i] = (_Float16)hn;
}

// ---------------- host launch ----------------
extern "C" void kernel_launch(void* const* d_in, const int* in_sizes, int n_in,
                              void* d_out, int out_size, void* d_ws, size_t ws_size,
                              hipStream_t stream) {
  (void)in_sizes; (void)n_in; (void)out_size; (void)ws_size;

  const int*   captions = (const int*)  d_in[0];
  const float* features = (const float*)d_in[1];
  const float* embed    = (const float*)d_in[2];
  const float* Vw = (const float*)d_in[3];  const float* Vb = (const float*)d_in[4];
  const float* Sw = (const float*)d_in[5];  const float* Sb = (const float*)d_in[6];
  const float* Uw = (const float*)d_in[7];  const float* Ub = (const float*)d_in[8];
  const float* Ww = (const float*)d_in[9];  const float* Wb = (const float*)d_in[10];
  const float* enc_w  = (const float*)d_in[11]; const float* enc_b  = (const float*)d_in[12];
  const float* dec_w  = (const float*)d_in[13]; const float* dec_b  = (const float*)d_in[14];
  const float* full_w = (const float*)d_in[15]; const float* full_b = (const float*)d_in[16];
  const float* fbeta_w= (const float*)d_in[17]; const float* fbeta_b= (const float*)d_in[18];
  const float* Cw = (const float*)d_in[19]; const float* Cb = (const float*)d_in[20];

  float* out    = (float*)d_out;                       // [T*B,VOCAB]
  float* alphas = out + (long)T_ * B_ * VOCAB_;        // [B,T,P]

  // ---- workspace carve ----
  char* ws = (char*)d_ws;
  size_t off = 0;
  auto take = [&](size_t bytes) -> char* {
    char* p = ws + off;
    off = (off + bytes + 255) & ~(size_t)255;
    return p;
  };
  const long BP = (long)B_ * P_;     // 12544
  _Float16* featH  = (_Float16*)take((size_t)BP * FEAT_ * 2);
  _Float16* encWH  = (_Float16*)take((size_t)FEAT_ * ATT_ * 2);
  _Float16* decWH  = (_Float16*)take((size_t)H_ * ATT_ * 2);
  _Float16* fbWH   = (_Float16*)take((size_t)H_ * FEAT_ * 2);
  _Float16* VwH    = (_Float16*)take((size_t)4 * IN_ * F_ * 2);
  _Float16* SwH    = (_Float16*)take((size_t)4 * F_ * F_ * 2);
  _Float16* UwH    = (_Float16*)take((size_t)4 * F_ * H_ * 2);
  _Float16* WwH    = (_Float16*)take((size_t)4 * H_ * H_ * 2);
  _Float16* CwH    = (_Float16*)take((size_t)H_ * VOCABP_ * 2);  // zero-padded cols
  _Float16* embsH  = (_Float16*)take((size_t)T_ * B_ * E_ * 2);
  float*    att1   = (float*)   take((size_t)BP * ATT_ * 4);
  float*    att2   = (float*)   take((size_t)B_ * ATT_ * 4);
  float*    awe    = (float*)   take((size_t)B_ * FEAT_ * 4);
  float*    gate   = (float*)   take((size_t)B_ * FEAT_ * 4);
  _Float16* xH     = (_Float16*)take((size_t)B_ * IN_ * 2);
  float*    g1F    = (float*)   take((size_t)4 * B_ * F_ * 4);
  _Float16* g1H    = (_Float16*)take((size_t)4 * B_ * F_ * 2);
  float*    g2F    = (float*)   take((size_t)4 * B_ * F_ * 4);
  _Float16* g2H    = (_Float16*)take((size_t)4 * B_ * F_ * 2);
  float*    g3F    = (float*)   take((size_t)4 * B_ * H_ * 4);
  float*    cSt    = (float*)   take((size_t)B_ * H_ * 4);
  _Float16* hH     = (_Float16*)take((size_t)B_ * H_ * 2);
  _Float16* hidH   = (_Float16*)take((size_t)T_ * B_ * H_ * 2);

  auto cvt = [&](const float* s, _Float16* d, long n) {
    int blocks = (int)((n + 255) / 256);
    if (blocks > 8192) blocks = 8192;
    f32_to_f16_kernel<<<blocks, 256, 0, stream>>>(s, d, n);
  };

  // ---- one-time: convert operands to f16, gather embeddings ----
  cvt(features, featH, BP * FEAT_);
  cvt(enc_w,  encWH, (long)FEAT_ * ATT_);
  cvt(dec_w,  decWH, (long)H_ * ATT_);
  cvt(fbeta_w,fbWH,  (long)H_ * FEAT_);
  cvt(Vw, VwH, 4L * IN_ * F_);
  cvt(Sw, SwH, 4L * F_ * F_);
  cvt(Uw, UwH, 4L * F_ * H_);
  cvt(Ww, WwH, 4L * H_ * H_);
  f32_to_f16_pad_kernel<<<8192, 256, 0, stream>>>(Cw, CwH, H_, VOCAB_, VOCABP_);
  gather_embs_kernel<<<(int)(((long)T_ * B_ * E_ + 255) / 256), 256, 0, stream>>>(
      captions, embed, embsH);

  // ---- att1 = features @ enc_w + enc_b : [12544,512] ----
  gemm_wmma_f16<0, false, false><<<dim3(ATT_ / TN, (int)(BP / TM), 1), 128, 0, stream>>>(
      featH, 0, encWH, 0, enc_b, 0, att1, 0, nullptr, 0, (int)BP, ATT_, FEAT_, ATT_);

  init_state_kernel<<<(B_ * H_ + 255) / 256, 256, 0, stream>>>(hH, cSt);

  // ---- sequential recurrence ----
  for (int t = 0; t < T_; ++t) {
    // att2 = h @ dec_w + dec_b  [64,512]
    gemm_wmma_f16<0, false, false><<<dim3(ATT_ / TN, B_ / TM, 1), 128, 0, stream>>>(
        hH, 0, decWH, 0, dec_b, 0, att2, 0, nullptr, 0, B_, ATT_, H_, ATT_);
    // scores + softmax + context (writes alphas straight into d_out)
    attention_kernel<<<B_, 256, 0, stream>>>(att1, att2, full_w, full_b,
                                             features, alphas, t, awe);
    // gate = sigmoid(h @ fbeta_w + fbeta_b)  [64,2048]
    gemm_wmma_f16<1, false, false><<<dim3(FEAT_ / TN, B_ / TM, 1), 128, 0, stream>>>(
        hH, 0, fbWH, 0, fbeta_b, 0, gate, 0, nullptr, 0, B_, FEAT_, H_, FEAT_);
    // x = concat(emb_t, gate*awe)  [64,2560] f16
    build_x_kernel<<<(int)(((long)B_ * IN_ + 255) / 256), 256, 0, stream>>>(
        embsH + (long)t * B_ * E_, gate, awe, xH);
    // g1 = x @ Vw[k] + Vb[k], k=0..3 (z-batched)
    gemm_wmma_f16<0, false, true><<<dim3(F_ / TN, B_ / TM, 4), 128, 0, stream>>>(
        xH, 0, VwH, (long)IN_ * F_, Vb, F_, g1F, (long)B_ * F_, g1H, (long)B_ * F_,
        B_, F_, IN_, F_);
    // g2 = g1 @ Sw[k] + Sb[k]
    gemm_wmma_f16<0, false, true><<<dim3(F_ / TN, B_ / TM, 4), 128, 0, stream>>>(
        g1H, (long)B_ * F_, SwH, (long)F_ * F_, Sb, F_, g2F, (long)B_ * F_,
        g2H, (long)B_ * F_, B_, F_, F_, F_);
    // g3 = g2 @ Uw[k] + Ub[k]
    gemm_wmma_f16<0, false, false><<<dim3(H_ / TN, B_ / TM, 4), 128, 0, stream>>>(
        g2H, (long)B_ * F_, UwH, (long)F_ * H_, Ub, H_, g3F, (long)B_ * H_,
        nullptr, 0, B_, H_, F_, H_);
    // g3 += h @ Ww[k] + Wb[k]
    gemm_wmma_f16<0, true, false><<<dim3(H_ / TN, B_ / TM, 4), 128, 0, stream>>>(
        hH, 0, WwH, (long)H_ * H_, Wb, H_, g3F, (long)B_ * H_,
        nullptr, 0, B_, H_, H_, H_);
    // pointwise LSTM: update c, h; stash h into hiddens[t]
    lstm_update_kernel<<<(B_ * H_ + 255) / 256, 256, 0, stream>>>(
        g3F, cSt, hH, hidH + (long)t * B_ * H_);
  }

  // ---- outputs = hiddens.reshape(T*B,H) @ CwH(padded) + Cb  [2560,20000] ----
  gemm_wmma_f16<0, false, false><<<dim3(VOCABP_ / TN, (T_ * B_) / TM, 1),
                                   128, 0, stream>>>(
      hidH, 0, CwH, 0, Cb, 0, out, 0, nullptr, 0, T_ * B_, VOCAB_, H_, VOCABP_);
}
